// Aspect_Attention_op2_80504866996811
// MI455X (gfx1250) — matrix operand order repaired and checked by reading, not compiled, not run on hardware
//
#include <hip/hip_runtime.h>
#include <hip/hip_bf16.h>
#include <math.h>

typedef __bf16 bf16;
typedef __attribute__((ext_vector_type(16))) __bf16 v16bf;
typedef __attribute__((ext_vector_type(8)))  float  v8f;
typedef __attribute__((ext_vector_type(4)))  int    v4i;

#define WMMA_BF16(a, b, c) \
  __builtin_amdgcn_wmma_f32_16x16x32_bf16(false, (a), false, (b), (short)0, (c), false, false)

#ifndef __has_builtin
#define __has_builtin(x) 0
#endif

// -------------------- CDNA5 async global->LDS copy helpers ---------------------
// GLOBAL_LOAD_ASYNC_TO_LDS_B128 (CDNA5 ISA op 98, CDNA5 ISA),
// completion tracked with ASYNCcnt -> S_WAIT_ASYNCCNT.
__device__ __forceinline__ void async_b128(const bf16* g, bf16* l) {
#if __has_builtin(__builtin_amdgcn_global_load_async_to_lds_b128)
  __builtin_amdgcn_global_load_async_to_lds_b128(
      (__attribute__((address_space(1))) v4i*)(unsigned long long)g,
      (__attribute__((address_space(3))) v4i*)(unsigned)(unsigned long long)l,
      0, 0);
#else
  // dsaddr comes from the low 32 bits of the flat LDS address (ISA 10.2 aperture rules)
  asm volatile("global_load_async_to_lds_b128 %0, %1, off"
               :: "v"((unsigned)(unsigned long long)l), "v"(g) : "memory");
#endif
}

__device__ __forceinline__ void wait_async() {
#if __has_builtin(__builtin_amdgcn_s_wait_asynccnt)
  __builtin_amdgcn_s_wait_asynccnt(0);
#else
  asm volatile("s_wait_asynccnt 0x0" ::: "memory");
#endif
}

// ---------- WMMA fragment loaders (contiguous k = 16*(lane/16) + i, same in A and B) ----

// A-matrix 16x32 bf16, source tile row-major with leading dim ld.
// lanes 0-15: M=lane, K=0..15; lanes 16-31: M=lane-16, K=16..31
// -> one contiguous 32B run per lane => 2x ds_load_b128.
__device__ __forceinline__ v16bf load_a_frag(const bf16* t, int ld) {
  int lane = threadIdx.x & 31;
  int hl = lane >> 4, m = lane & 15;
  v16bf a;
#pragma unroll
  for (int i = 0; i < 16; ++i) {
    int k = (hl << 4) + i;
    a[i] = t[m * ld + k];
  }
  return a;
}

// B-matrix 32x16 bf16 from transposed source: b[k][n] = t[n*ld + k]
// (per lane: 16 contiguous bf16 = one 32B run => 2x ds_load_b128)
__device__ __forceinline__ v16bf load_b_fragT(const bf16* t, int ld) {
  int lane = threadIdx.x & 31;
  int hl = lane >> 4, n = lane & 15;
  v16bf b;
#pragma unroll
  for (int i = 0; i < 16; ++i) {
    int k = (hl << 4) + i;
    b[i] = t[n * ld + k];
  }
  return b;
}

// ---------------- Kernel 1: weights[b,l] = att_v . tanh(x2 @ att_W) -------------
// 2048 blocks x 256 thr (8 waves). att_W K-band staged transposed in LDS so
// B-fragments are contiguous ds_load_b128 instead of 16 scattered global b32.
__global__ __launch_bounds__(256) void weights_kernel(
    const float* __restrict__ x2, const float* __restrict__ attW,
    const float* __restrict__ attv, float* __restrict__ wout) {
  __shared__ bf16  atile[16 * 32];
  __shared__ bf16  bandT[768 * 40];  // attW[kc*32+k][c] -> bandT[c*40+k] (pad 32->40)
  __shared__ float partial[16];
  const int row0 = blockIdx.x * 16;
  const int tid = threadIdx.x;
  const int wv = tid >> 5, lane = tid & 31, hl = lane >> 4;
  if (tid < 16) partial[tid] = 0.f;

  const v8f vzero = {0.f, 0.f, 0.f, 0.f, 0.f, 0.f, 0.f, 0.f};
  v8f acc[6];
#pragma unroll
  for (int j = 0; j < 6; ++j) acc[j] = vzero;

  for (int kc = 0; kc < 24; ++kc) {
    __syncthreads();
#pragma unroll
    for (int e = tid; e < 512; e += 256) {
      int m = e >> 5, k = e & 31;
      atile[e] = (bf16)x2[(size_t)(row0 + m) * 768 + kc * 32 + k];
    }
    for (int k = 0; k < 32; ++k)
      for (int c = tid; c < 768; c += 256)
        bandT[c * 40 + k] = (bf16)attW[(size_t)(kc * 32 + k) * 768 + c];
    __syncthreads();
    v16bf a = load_a_frag(atile, 32);
#pragma unroll
    for (int j = 0; j < 6; ++j) {
      int nt = wv + 8 * j;
      v16bf bb = load_b_fragT(bandT + nt * 16 * 40, 40);
      acc[j] = WMMA_BF16(a, bb, acc[j]);
    }
  }
  __syncthreads();
  // epilogue: partial[m] += sum_n tanh(acc[m][n]) * att_v[n]
#pragma unroll
  for (int j = 0; j < 6; ++j) {
    int nt = wv + 8 * j;
    float av = attv[nt * 16 + (lane & 15)];
#pragma unroll
    for (int r = 0; r < 8; ++r) {
      float v = tanhf(acc[j][r]) * av;
      v += __shfl_xor(v, 1, 32);
      v += __shfl_xor(v, 2, 32);
      v += __shfl_xor(v, 4, 32);
      v += __shfl_xor(v, 8, 32);
      if ((lane & 15) == 0) atomicAdd(&partial[r + hl * 8], v);
    }
  }
  __syncthreads();
  if (tid < 16) wout[row0 + tid] = partial[tid];
}

// ---------------- Kernel 2: softmax over batch axis (B=16) per l ----------------
__global__ __launch_bounds__(256) void softmax_b_kernel(
    const float* __restrict__ w, float* __restrict__ wsm) {
  int l = blockIdx.x * 256 + threadIdx.x;
  if (l >= 2048) return;
  float m = -__builtin_inff();
#pragma unroll
  for (int b = 0; b < 16; ++b) m = fmaxf(m, w[b * 2048 + l]);
  float s = 0.f, e[16];
#pragma unroll
  for (int b = 0; b < 16; ++b) { e[b] = __expf(w[b * 2048 + l] - m); s += e[b]; }
  float inv = 1.f / s;
#pragma unroll
  for (int b = 0; b < 16; ++b) wsm[b * 2048 + l] = e[b] * inv;
}

// --------- Kernel 2.5: one-time bf16 conversion + weight modulation ------------
// xb = bf16(x); vb[b,l,d] = bf16(x2[b,l,d] * wsm[b, (l*768+d) mod 2048])
__global__ __launch_bounds__(256) void prep_kernel(
    const float* __restrict__ x, const float* __restrict__ x2,
    const float* __restrict__ wsm, bf16* __restrict__ xb, bf16* __restrict__ vb) {
  const size_t N = (size_t)16 * 2048 * 768;
  const size_t stride = (size_t)gridDim.x * 256;
  for (size_t e = (size_t)blockIdx.x * 256 + threadIdx.x; e < N; e += stride) {
    xb[e] = (bf16)x[e];
    size_t b = e / 1572864, rem = e - b * 1572864;   // rem = l*768+d
    vb[e] = (bf16)(x2[e] * wsm[b * 2048 + (rem & 2047)]);
  }
}

// --------- Kernel 2.6: tiled transpose vb[b][l][d] -> vT[b][d][l] --------------
__global__ __launch_bounds__(256) void transpose_kernel(
    const bf16* __restrict__ vb, bf16* __restrict__ vT) {
  __shared__ bf16 tile[64][65];
  const int b = blockIdx.z, l0 = blockIdx.x * 64, d0 = blockIdx.y * 64;
  const int c = threadIdx.x & 63, rg = threadIdx.x >> 6;
#pragma unroll
  for (int rr = 0; rr < 16; ++rr) {
    int r = rg * 16 + rr;
    tile[r][c] = vb[(size_t)(b * 2048 + l0 + r) * 768 + d0 + c];
  }
  __syncthreads();
#pragma unroll
  for (int rr = 0; rr < 16; ++rr) {
    int r = rg * 16 + rr;
    vT[(size_t)(b * 768 + d0 + r) * 2048 + l0 + c] = tile[c][r];
  }
}

// ---------------- Kernel 3A: flash attention, async-staged bf16 ----------------
// Grid (64 q-tiles of 32, 16 batches), 256 thr = 8 waves.
// All staging is GLOBAL_LOAD_ASYNC_TO_LDS_B128 (no VGPR round trip, no cvt).
// Dynamic LDS:
//   qs  bf16[32*768] @0       (49152)   query tile, row-major
//   ks  bf16[64*768] @49152   (98304)   V chunk, key-major (score phase)
//   ksT bf16[768*72] @147456  (110592)  V chunk, d-major pad 64->72 (context phase)
//   S   f32 [32*64]  @258048  (8192)
//   P   bf16[32*64]  @266240  (4096)
//   stats f32[96]    @270336  (384)  -> total 270720
__global__ __launch_bounds__(256) void flash_async_kernel(
    const bf16* __restrict__ xb, const bf16* __restrict__ vb,
    const bf16* __restrict__ vT, float* __restrict__ out) {
  extern __shared__ char smem[];
  bf16*  qs     = (bf16*)smem;
  bf16*  ks     = (bf16*)(smem + 49152);
  bf16*  ksT    = (bf16*)(smem + 147456);
  float* S      = (float*)(smem + 258048);
  bf16*  P      = (bf16*)(smem + 266240);
  float* rowmax = (float*)(smem + 270336);
  float* rowsum = rowmax + 32;
  float* fact   = rowmax + 64;

  const int tid = threadIdx.x;
  const int wv = tid >> 5, lane = tid & 31, hl = lane >> 4;
  const int bb = blockIdx.y;
  const int q0 = blockIdx.x * 32;
  const int mt = wv & 1;    // query 16-row subtile
  const int nt = wv >> 1;   // score key-column tile (0..3)
  const int dblk = wv >> 1; // context d-block (0..3), 12 d-tiles each

  // async-stage Q tile: one contiguous 49152B span
  {
    const bf16* qg = xb + (size_t)(bb * 2048 + q0) * 768;
#pragma unroll
    for (int j = 0; j < 12; ++j) {
      int t = tid + j * 256;
      async_b128(qg + t * 8, qs + t * 8);
    }
  }
  if (tid < 32) { rowmax[tid] = -__builtin_inff(); rowsum[tid] = 0.f; }

  const v8f vzero = {0.f, 0.f, 0.f, 0.f, 0.f, 0.f, 0.f, 0.f};
  v8f acc[12];
#pragma unroll
  for (int t = 0; t < 12; ++t) acc[t] = vzero;

  for (int kc = 0; kc < 32; ++kc) {
    const int k0 = kc * 64;
    __syncthreads();  // previous chunk fully consumed
    // async-stage V chunk, key-major (contiguous 98304B)
    const bf16* vg = vb + (size_t)(bb * 2048 + k0) * 768;
#pragma unroll
    for (int j = 0; j < 24; ++j) {
      int t = tid + j * 256;
      async_b128(vg + t * 8, ks + t * 8);
    }
    // async-stage V chunk, d-major: 768 rows x 128B from vT
    const bf16* vTg = vT + (size_t)bb * 768 * 2048 + k0;
#pragma unroll
    for (int j = 0; j < 24; ++j) {
      int t = tid + j * 256;
      int d = t >> 3, seg = t & 7;
      async_b128(vTg + (size_t)d * 2048 + seg * 8, ksT + d * 72 + seg * 8);
    }
    wait_async();
    __syncthreads();
    // S = Q . V^T (16x16 tile per wave, 24 K-steps over D=768)
    {
      v8f s = vzero;
      for (int dc = 0; dc < 24; ++dc) {
        v16bf a  = load_a_frag (qs + mt * 16 * 768 + dc * 32, 768);
        v16bf bt = load_b_fragT(ks + nt * 16 * 768 + dc * 32, 768);
        s = WMMA_BF16(a, bt, s);
      }
#pragma unroll
      for (int r = 0; r < 8; ++r)
        S[(mt * 16 + r + hl * 8) * 64 + nt * 16 + (lane & 15)] = s[r];
    }
    __syncthreads();
    // online softmax (one thread per query row)
    if (tid < 32) {
      const int row = tid;
      float mold = rowmax[row], mnew = mold;
      for (int j = 0; j < 64; ++j) mnew = fmaxf(mnew, S[row * 64 + j]);
      float f = __expf(mold - mnew);
      float ls = rowsum[row] * f;
      for (int j = 0; j < 64; ++j) {
        float p = __expf(S[row * 64 + j] - mnew);
        P[row * 64 + j] = (bf16)p;
        ls += p;
      }
      rowmax[row] = mnew; rowsum[row] = ls; fact[row] = f;
    }
    __syncthreads();
    // rescale accumulators
    float fr[8];
#pragma unroll
    for (int r = 0; r < 8; ++r) fr[r] = fact[mt * 16 + r + hl * 8];
#pragma unroll
    for (int t = 0; t < 12; ++t)
#pragma unroll
      for (int r = 0; r < 8; ++r) acc[t][r] *= fr[r];
    // context += P(16x64) . V ; B-frags from d-major ksT => contiguous 32B runs
    v16bf a0 = load_a_frag(P + mt * 16 * 64, 64);
    v16bf a1 = load_a_frag(P + mt * 16 * 64 + 32, 64);
#pragma unroll
    for (int t = 0; t < 12; ++t) {
      const int dt = dblk * 12 + t;
      v16bf b0 = load_b_fragT(ksT + dt * 16 * 72, 72);       // keys 0..31
      v16bf b1 = load_b_fragT(ksT + dt * 16 * 72 + 32, 72);  // keys 32..63
      acc[t] = WMMA_BF16(a0, b0, acc[t]);
      acc[t] = WMMA_BF16(a1, b1, acc[t]);
    }
  }
  __syncthreads();
  float invr[8];
#pragma unroll
  for (int r = 0; r < 8; ++r) invr[r] = 1.f / rowsum[mt * 16 + r + hl * 8];
#pragma unroll
  for (int t = 0; t < 12; ++t) {
    const int dt = dblk * 12 + t;
#pragma unroll
    for (int r = 0; r < 8; ++r) {
      const int qrow = q0 + mt * 16 + r + hl * 8;
      out[(size_t)(bb * 2048 + qrow) * 768 + dt * 16 + (lane & 15)] =
          acc[t][r] * invr[r];
    }
  }
}

// ---------------- Kernel 3B: fallback (on-the-fly f32->bf16) -------------------
__global__ __launch_bounds__(256) void flash_kernel(
    const float* __restrict__ x, const float* __restrict__ x2,
    const float* __restrict__ wsm, float* __restrict__ out) {
  extern __shared__ char smem[];
  float* wrow   = (float*)smem;
  bf16*  qs     = (bf16*)(smem + 8192);
  bf16*  ks     = (bf16*)(smem + 57344);
  float* S      = (float*)(smem + 155648);
  bf16*  P      = (bf16*)(smem + 163840);
  float* rowmax = (float*)(smem + 167936);
  float* rowsum = rowmax + 32;
  float* fact   = rowmax + 64;

  const int tid = threadIdx.x;
  const int wv = tid >> 5, lane = tid & 31, hl = lane >> 4;
  const int bb = blockIdx.y;
  const int q0 = blockIdx.x * 32;
  const int mt = wv & 1, nt = wv >> 1, dblk = wv >> 1;

  for (int i = tid; i < 2048; i += 256) wrow[i] = wsm[bb * 2048 + i];
  for (int m = 0; m < 32; ++m)
    for (int d = tid; d < 768; d += 256)
      qs[m * 768 + d] = (bf16)x[(size_t)(bb * 2048 + q0 + m) * 768 + d];
  if (tid < 32) { rowmax[tid] = -__builtin_inff(); rowsum[tid] = 0.f; }

  const v8f vzero = {0.f, 0.f, 0.f, 0.f, 0.f, 0.f, 0.f, 0.f};
  v8f acc[12];
#pragma unroll
  for (int t = 0; t < 12; ++t) acc[t] = vzero;

  for (int kc = 0; kc < 32; ++kc) {
    const int k0 = kc * 64;
    __syncthreads();
    for (int kr = 0; kr < 64; ++kr) {
      const int krow = k0 + kr;
      const float* src = x2 + (size_t)(bb * 2048 + krow) * 768;
      const int base = (krow * 768) & 2047;
      for (int d = tid; d < 768; d += 256)
        ks[kr * 768 + d] = (bf16)(src[d] * wrow[(base + d) & 2047]);
    }
    __syncthreads();
    {
      v8f s = vzero;
      for (int dc = 0; dc < 24; ++dc) {
        v16bf a  = load_a_frag (qs + mt * 16 * 768 + dc * 32, 768);
        v16bf bt = load_b_fragT(ks + nt * 16 * 768 + dc * 32, 768);
        s = WMMA_BF16(a, bt, s);
      }
#pragma unroll
      for (int r = 0; r < 8; ++r)
        S[(mt * 16 + r + hl * 8) * 64 + nt * 16 + (lane & 15)] = s[r];
    }
    __syncthreads();
    if (tid < 32) {
      const int row = tid;
      float mold = rowmax[row], mnew = mold;
      for (int j = 0; j < 64; ++j) mnew = fmaxf(mnew, S[row * 64 + j]);
      float f = __expf(mold - mnew);
      float ls = rowsum[row] * f;
      for (int j = 0; j < 64; ++j) {
        float p = __expf(S[row * 64 + j] - mnew);
        P[row * 64 + j] = (bf16)p;
        ls += p;
      }
      rowmax[row] = mnew; rowsum[row] = ls; fact[row] = f;
    }
    __syncthreads();
    float fr[8];
#pragma unroll
    for (int r = 0; r < 8; ++r) fr[r] = fact[mt * 16 + r + hl * 8];
#pragma unroll
    for (int t = 0; t < 12; ++t)
#pragma unroll
      for (int r = 0; r < 8; ++r) acc[t][r] *= fr[r];
    v16bf a0 = load_a_frag(P + mt * 16 * 64, 64);
    v16bf a1 = load_a_frag(P + mt * 16 * 64 + 32, 64);
#pragma unroll
    for (int t = 0; t < 12; ++t) {
      const int dt = dblk * 12 + t;
      // strided fallback reads (no d-major copy in this variant)
      int n = lane & 15;
      v16bf b0, b1;
#pragma unroll
      for (int i = 0; i < 16; ++i) {
        int k = (hl << 4) + i;
        b0[i] = ks[k * 768 + dt * 16 + n];
        b1[i] = ks[(k + 32) * 768 + dt * 16 + n];
      }
      acc[t] = WMMA_BF16(a0, b0, acc[t]);
      acc[t] = WMMA_BF16(a1, b1, acc[t]);
    }
  }
  __syncthreads();
  float invr[8];
#pragma unroll
  for (int r = 0; r < 8; ++r) invr[r] = 1.f / rowsum[mt * 16 + r + hl * 8];
#pragma unroll
  for (int t = 0; t < 12; ++t) {
    const int dt = dblk * 12 + t;
#pragma unroll
    for (int r = 0; r < 8; ++r) {
      const int qrow = q0 + mt * 16 + r + hl * 8;
      out[(size_t)(bb * 2048 + qrow) * 768 + dt * 16 + (lane & 15)] =
          acc[t][r] * invr[r];
    }
  }
}

// ------------------------------- launcher --------------------------------------
extern "C" void kernel_launch(void* const* d_in, const int* in_sizes, int n_in,
                              void* d_out, int out_size, void* d_ws, size_t ws_size,
                              hipStream_t stream) {
  (void)in_sizes; (void)n_in; (void)out_size;
  const float* x     = (const float*)d_in[0];   // [16,2048,768]
  const float* x2    = (const float*)d_in[1];   // [16,2048,768]
  const float* att_v = (const float*)d_in[2];   // [768]
  const float* att_W = (const float*)d_in[3];   // [768,768]
  float* out = (float*)d_out;                   // [16,2048,768] f32

  float* w_raw = (float*)d_ws;                  // 32768 f32
  float* w_sm  = w_raw + 16 * 2048;             // 32768 f32
  const size_t NE = (size_t)16 * 2048 * 768;
  const size_t base = 262144;
  const size_t need = base + 3 * NE * sizeof(bf16);  // xb + vb + vT ~ 151 MB

  weights_kernel<<<2048, 256, 0, stream>>>(x2, att_W, att_v, w_raw);
  softmax_b_kernel<<<8, 256, 0, stream>>>(w_raw, w_sm);

  if (ws_size >= need) {
    bf16* xb = (bf16*)((char*)d_ws + base);
    bf16* vb = xb + NE;
    bf16* vT = vb + NE;
    prep_kernel<<<4096, 256, 0, stream>>>(x, x2, w_sm, xb, vb);
    transpose_kernel<<<dim3(32, 12, 16), 256, 0, stream>>>(vb, vT);
    flash_async_kernel<<<dim3(64, 16), 256, 270720, stream>>>(xb, vb, vT, out);
  } else {
    flash_kernel<<<dim3(64, 16), 256, 168320, stream>>>(x, x2, w_sm, out);
  }
}
